// MultiScaleAttention_70617852281009
// MI455X (gfx1250) — matrix operand order repaired and checked by reading, not compiled, not run on hardware
//
#include <hip/hip_runtime.h>
#include <hip/hip_bf16.h>
#include <math.h>

typedef __attribute__((ext_vector_type(16))) __bf16 v16bf;
typedef __attribute__((ext_vector_type(8)))  __bf16 v8bf;
typedef __attribute__((ext_vector_type(4)))  __bf16 v4bf;
typedef __attribute__((ext_vector_type(8)))  float  v8f;

constexpr int Dd = 1024, Hh = 16, DH = 64, Bb = 4, Ss = 2048;

__device__ __forceinline__ v16bf cat16(v8bf lo, v8bf hi) {
  return __builtin_shufflevector(lo, hi, 0, 1, 2, 3, 4, 5, 6, 7, 8, 9, 10, 11,
                                 12, 13, 14, 15);
}

// ---------------------------------------------------------------------------
// Kernel 0: fp32 -> bf16 convert (Wo), 8 elems/thread, vectorized.
// ---------------------------------------------------------------------------
__global__ __launch_bounds__(256) void cvt_kernel(const float* __restrict__ src,
                                                  __bf16* __restrict__ dst, int n) {
  int i = (blockIdx.x * 256 + threadIdx.x) * 8;
  if (i + 8 <= n) {
    float4 a = *(const float4*)(src + i);
    float4 b = *(const float4*)(src + i + 4);
    v8bf t;
    t[0] = (__bf16)a.x; t[1] = (__bf16)a.y; t[2] = (__bf16)a.z; t[3] = (__bf16)a.w;
    t[4] = (__bf16)b.x; t[5] = (__bf16)b.y; t[6] = (__bf16)b.z; t[7] = (__bf16)b.w;
    *(v8bf*)(dst + i) = t;
  }
}

// ---------------------------------------------------------------------------
// Kernel 1: per-head QKV projection.  grid = (S/32, B*H, 3), block = 256.
// K-step 64: 2 WMMAs per barrier pair per wave.
// ---------------------------------------------------------------------------
__global__ __launch_bounds__(256) void qkv_kernel(
    const float* __restrict__ x,
    const float* __restrict__ Wq, const float* __restrict__ bq,
    const float* __restrict__ Wk, const float* __restrict__ bk,
    const float* __restrict__ Wv, const float* __restrict__ bv,
    __bf16* __restrict__ qo, __bf16* __restrict__ ko, __bf16* __restrict__ vo) {
  int which = blockIdx.z;
  const float* W    = (which == 0) ? Wq : (which == 1) ? Wk : Wv;
  const float* bias = (which == 0) ? bq : (which == 1) ? bk : bv;
  __bf16* out       = (which == 0) ? qo : (which == 1) ? ko : vo;

  int bh = blockIdx.y;
  int h = bh & 15;
  int s0 = blockIdx.x * 32;
  int tid = threadIdx.x;
  int wv = tid >> 5, lane = tid & 31;
  int hs = lane >> 4, ln = lane & 15;
  int mt = wv >> 2, nt = wv & 3;

  __shared__ alignas(16) __bf16 As[32][72];    // [m][k], pitch 144B
  __shared__ alignas(16) __bf16 BsT[64][72];   // [n][k] transposed
  __shared__ alignas(16) __bf16 Outs[32][72];  // epilogue staging

  const float* xrow  = x + ((size_t)(bh >> 4) * Ss + s0) * Dd;
  const float* wbase = W + (size_t)h * Dd * DH;

  // Static indices for staging (straight-line, no dynamic trip counts).
  const int as_r = tid >> 4, as_c = (tid & 15) * 4;          // A: 2 tasks
  const int wt_k = (tid >> 4) * 4, wt_c = (tid & 15) * 4;    // W: 4x4 transpose

  v8f acc = {};
  for (int k0 = 0; k0 < Dd; k0 += 64) {
    // Stage X tile 32x64 fp32->bf16 (b64 LDS stores), 2 tasks/thread.
#pragma unroll
    for (int it = 0; it < 2; ++it) {
      int r = as_r + it * 16;
      float4 f = *(const float4*)(xrow + (size_t)r * Dd + k0 + as_c);
      v4bf t;
      t[0] = (__bf16)f.x; t[1] = (__bf16)f.y; t[2] = (__bf16)f.z; t[3] = (__bf16)f.w;
      *(v4bf*)&As[r][as_c] = t;
    }
    // Stage W tile 64x64 transposed via per-thread 4x4 register transpose.
    {
      float4 f0 = *(const float4*)(wbase + (size_t)(k0 + wt_k + 0) * DH + wt_c);
      float4 f1 = *(const float4*)(wbase + (size_t)(k0 + wt_k + 1) * DH + wt_c);
      float4 f2 = *(const float4*)(wbase + (size_t)(k0 + wt_k + 2) * DH + wt_c);
      float4 f3 = *(const float4*)(wbase + (size_t)(k0 + wt_k + 3) * DH + wt_c);
      v4bf t0, t1, t2, t3;
      t0[0] = (__bf16)f0.x; t0[1] = (__bf16)f1.x; t0[2] = (__bf16)f2.x; t0[3] = (__bf16)f3.x;
      t1[0] = (__bf16)f0.y; t1[1] = (__bf16)f1.y; t1[2] = (__bf16)f2.y; t1[3] = (__bf16)f3.y;
      t2[0] = (__bf16)f0.z; t2[1] = (__bf16)f1.z; t2[2] = (__bf16)f2.z; t2[3] = (__bf16)f3.z;
      t3[0] = (__bf16)f0.w; t3[1] = (__bf16)f1.w; t3[2] = (__bf16)f2.w; t3[3] = (__bf16)f3.w;
      *(v4bf*)&BsT[wt_c + 0][wt_k] = t0;
      *(v4bf*)&BsT[wt_c + 1][wt_k] = t1;
      *(v4bf*)&BsT[wt_c + 2][wt_k] = t2;
      *(v4bf*)&BsT[wt_c + 3][wt_k] = t3;
    }
    __syncthreads();

    const __bf16* ar = &As[mt * 16 + ln][0];
    const __bf16* br = &BsT[nt * 16 + ln][0];
#pragma unroll
    for (int ks = 0; ks < 2; ++ks) {
      v16bf a = cat16(*(const v8bf*)(ar + ks * 32 + hs * 8),
                      *(const v8bf*)(ar + ks * 32 + 16 + hs * 8));
      v16bf bfr = cat16(*(const v8bf*)(br + ks * 32 + hs * 8),
                        *(const v8bf*)(br + ks * 32 + 16 + hs * 8));
      acc = __builtin_amdgcn_wmma_f32_16x16x32_bf16(false, a, false, bfr,
                                                    (short)0, acc, false, false);
    }
    __syncthreads();
  }

  // Epilogue: bias add, stage to LDS, coalesced b128 stores (1 task/thread).
#pragma unroll
  for (int r = 0; r < 8; ++r) {
    int m = mt * 16 + r + 8 * hs;
    int n = nt * 16 + ln;
    Outs[m][n] = (__bf16)(acc[r] + bias[h * DH + n]);
  }
  __syncthreads();
  {
    __bf16* obase = out + ((size_t)bh * Ss + s0) * DH;
    int r = tid >> 3, ch = (tid & 7) * 8;
    *(v8bf*)(obase + (size_t)r * DH + ch) = *(const v8bf*)&Outs[r][ch];
  }
}

// ---------------------------------------------------------------------------
// Kernel 2: flash attention per (b,h).  grid = (S/128, B*H), block = 256.
// ---------------------------------------------------------------------------
__global__ __launch_bounds__(256) void attn_kernel(
    const __bf16* __restrict__ q, const __bf16* __restrict__ k,
    const __bf16* __restrict__ v, __bf16* __restrict__ concat) {
  int bh = blockIdx.y;
  int b = bh >> 4, h = bh & 15;
  int tid = threadIdx.x, wv = tid >> 5, lane = tid & 31;
  int hs = lane >> 4, ln = lane & 15;
  int s0 = blockIdx.x * 128 + wv * 16;

  __shared__ alignas(16) __bf16 Ks[64][72];       // [t][e] row-major
  __shared__ alignas(16) __bf16 VsT[64][72];      // [e][t] transposed
  __shared__ alignas(16) __bf16 Ps[8][16][72];    // per-wave P / out staging

  const __bf16* qb = q + ((size_t)bh * Ss) * DH;
  const __bf16* kb = k + ((size_t)bh * Ss) * DH;
  const __bf16* vb = v + ((size_t)bh * Ss) * DH;

  // Static staging indices.
  const int ks_r = tid >> 3, ks_ch = (tid & 7) * 8;        // K copy: 2 tasks
  const int vt_r = (tid >> 4) * 4, vt_e = (tid & 15) * 4;  // V: 4x4 transpose

  // Q fragments (global b128 loads), fold in 1/sqrt(DH) = 0.125.
  v16bf qa[2];
  {
    const __bf16* qrow = qb + (size_t)(s0 + ln) * DH;
#pragma unroll
    for (int ks = 0; ks < 2; ++ks) {
      v16bf t = cat16(*(const v8bf*)(qrow + ks * 32 + hs * 8),
                      *(const v8bf*)(qrow + ks * 32 + 16 + hs * 8));
#pragma unroll
      for (int i = 0; i < 16; ++i)
        qa[ks][i] = (__bf16)((float)t[i] * 0.125f);
    }
  }

  float m_i[8], l_i[8];
#pragma unroll
  for (int r = 0; r < 8; ++r) { m_i[r] = -1e30f; l_i[r] = 0.f; }
  v8f o[4] = {};

  for (int t0 = 0; t0 < Ss; t0 += 64) {
    // Stage K row-major (b128 copies, 2 tasks/thread).
#pragma unroll
    for (int it = 0; it < 2; ++it) {
      int r = ks_r + it * 32;
      *(v8bf*)&Ks[r][ks_ch] = *(const v8bf*)(kb + (size_t)(t0 + r) * DH + ks_ch);
    }
    // Stage V transposed via per-thread 4x4 register transpose.
    {
      v4bf a0 = *(const v4bf*)(vb + (size_t)(t0 + vt_r + 0) * DH + vt_e);
      v4bf a1 = *(const v4bf*)(vb + (size_t)(t0 + vt_r + 1) * DH + vt_e);
      v4bf a2 = *(const v4bf*)(vb + (size_t)(t0 + vt_r + 2) * DH + vt_e);
      v4bf a3 = *(const v4bf*)(vb + (size_t)(t0 + vt_r + 3) * DH + vt_e);
#pragma unroll
      for (int j = 0; j < 4; ++j) {
        v4bf t;
        t[0] = a0[j]; t[1] = a1[j]; t[2] = a2[j]; t[3] = a3[j];
        *(v4bf*)&VsT[vt_e + j][vt_r] = t;
      }
    }
    __syncthreads();

    // S = (Q*0.125) K^T : 16 x 64 per wave.
    v8f sc[4] = {};
#pragma unroll
    for (int nt = 0; nt < 4; ++nt) {
      const __bf16* kr = &Ks[nt * 16 + ln][0];
#pragma unroll
      for (int ks = 0; ks < 2; ++ks) {
        v16bf bbf = cat16(*(const v8bf*)(kr + ks * 32 + hs * 8),
                          *(const v8bf*)(kr + ks * 32 + 16 + hs * 8));
        sc[nt] = __builtin_amdgcn_wmma_f32_16x16x32_bf16(
            false, qa[ks], false, bbf, (short)0, sc[nt], false, false);
      }
    }

    // Online softmax over this 64-wide key block.
    float corr[8];
#pragma unroll
    for (int r = 0; r < 8; ++r) {
      float mx = fmaxf(fmaxf(sc[0][r], sc[1][r]), fmaxf(sc[2][r], sc[3][r]));
#pragma unroll
      for (int off = 1; off < 16; off <<= 1)
        mx = fmaxf(mx, __shfl_xor(mx, off, 16));
      float mnew = fmaxf(m_i[r], mx);
      corr[r] = __expf(m_i[r] - mnew);
      float ps = 0.f;
#pragma unroll
      for (int nt = 0; nt < 4; ++nt) {
        float p = __expf(sc[nt][r] - mnew);
        sc[nt][r] = p;
        ps += p;
      }
#pragma unroll
      for (int off = 1; off < 16; off <<= 1)
        ps += __shfl_xor(ps, off, 16);
      l_i[r] = l_i[r] * corr[r] + ps;
      m_i[r] = mnew;
    }

    // Re-swizzle P (C layout -> A layout) through this wave's LDS tile.
#pragma unroll
    for (int nt = 0; nt < 4; ++nt)
#pragma unroll
      for (int r = 0; r < 8; ++r)
        Ps[wv][r + 8 * hs][nt * 16 + ln] = (__bf16)sc[nt][r];
    asm volatile("s_wait_dscnt 0" ::: "memory");

#pragma unroll
    for (int nt = 0; nt < 4; ++nt)
#pragma unroll
      for (int r = 0; r < 8; ++r) o[nt][r] *= corr[r];

    const __bf16* pr = &Ps[wv][ln][0];
    v16bf pa[2];
#pragma unroll
    for (int ks = 0; ks < 2; ++ks)
      pa[ks] = cat16(*(const v8bf*)(pr + ks * 32 + hs * 8),
                     *(const v8bf*)(pr + ks * 32 + 16 + hs * 8));
#pragma unroll
    for (int nt = 0; nt < 4; ++nt) {
      const __bf16* vr = &VsT[nt * 16 + ln][0];
#pragma unroll
      for (int ks = 0; ks < 2; ++ks) {
        v16bf vbf = cat16(*(const v8bf*)(vr + ks * 32 + hs * 8),
                          *(const v8bf*)(vr + ks * 32 + 16 + hs * 8));
        o[nt] = __builtin_amdgcn_wmma_f32_16x16x32_bf16(
            false, pa[ks], false, vbf, (short)0, o[nt], false, false);
      }
    }
    __syncthreads();
  }

  // Epilogue: normalize, stage 16x64 tile in LDS, coalesced b128 stores.
#pragma unroll
  for (int nt = 0; nt < 4; ++nt)
#pragma unroll
    for (int r = 0; r < 8; ++r)
      Ps[wv][r + 8 * hs][nt * 16 + ln] = (__bf16)(o[nt][r] / l_i[r]);
  asm volatile("s_wait_dscnt 0" ::: "memory");
  {
    __bf16* cbase = concat + ((size_t)b * Ss + s0) * Dd + h * DH;
#pragma unroll
    for (int it = 0; it < 4; ++it) {
      int i = lane + it * 32;
      int r = i >> 3, ch = (i & 7) * 8;
      *(v8bf*)(cbase + (size_t)r * Dd + ch) = *(const v8bf*)&Ps[wv][r][ch];
    }
  }
}

// ---------------------------------------------------------------------------
// Kernel 3: y = concat @ Wo + bo + x.  grid = (B*S/32, D/128), block = 256.
// K-step 64: 4 WMMAs per barrier pair per wave.
// ---------------------------------------------------------------------------
__global__ __launch_bounds__(256) void oproj_kernel(
    const __bf16* __restrict__ concat, const __bf16* __restrict__ woh,
    const float* __restrict__ bo, const float* __restrict__ x,
    float* __restrict__ y) {
  int row0 = blockIdx.x * 32;
  int n0 = blockIdx.y * 128;
  int tid = threadIdx.x, wv = tid >> 5, lane = tid & 31;
  int hs = lane >> 4, ln = lane & 15;

  __shared__ alignas(16) __bf16 As[32][72];     // [m][k]
  __shared__ alignas(16) __bf16 BsT[128][72];   // [n][k] transposed

  const int as_r = tid >> 3, as_ch = (tid & 7) * 8;          // A copy: 1 task
  const int wt_k = (tid >> 5) * 4, wt_c = (tid & 31) * 4;    // Wo: 4x4, 2 tasks

  v8f acc[2] = {};
  for (int k0 = 0; k0 < Dd; k0 += 64) {
    // A tile 32x64: b128 copies.
    *(v8bf*)&As[as_r][as_ch] =
        *(const v8bf*)(concat + (size_t)(row0 + as_r) * Dd + k0 + as_ch);
    // Wo tile 64x128 transposed via per-thread 4x4 register transpose.
#pragma unroll
    for (int it = 0; it < 2; ++it) {
      int kk = wt_k + it * 32;
      v4bf a0 = *(const v4bf*)(woh + (size_t)(k0 + kk + 0) * Dd + n0 + wt_c);
      v4bf a1 = *(const v4bf*)(woh + (size_t)(k0 + kk + 1) * Dd + n0 + wt_c);
      v4bf a2 = *(const v4bf*)(woh + (size_t)(k0 + kk + 2) * Dd + n0 + wt_c);
      v4bf a3 = *(const v4bf*)(woh + (size_t)(k0 + kk + 3) * Dd + n0 + wt_c);
#pragma unroll
      for (int j = 0; j < 4; ++j) {
        v4bf t;
        t[0] = a0[j]; t[1] = a1[j]; t[2] = a2[j]; t[3] = a3[j];
        *(v4bf*)&BsT[wt_c + j][kk] = t;
      }
    }
    if (k0 + 64 < Dd)
      __builtin_prefetch(&woh[(size_t)(k0 + 64) * Dd + n0 + tid], 0, 1);
    __syncthreads();

    const __bf16* br = &BsT[wv * 16 + ln][0];
#pragma unroll
    for (int ks = 0; ks < 2; ++ks) {
      v16bf bbf = cat16(*(const v8bf*)(br + ks * 32 + hs * 8),
                        *(const v8bf*)(br + ks * 32 + 16 + hs * 8));
#pragma unroll
      for (int mt = 0; mt < 2; ++mt) {
        const __bf16* ar = &As[mt * 16 + ln][0];
        v16bf a = cat16(*(const v8bf*)(ar + ks * 32 + hs * 8),
                        *(const v8bf*)(ar + ks * 32 + 16 + hs * 8));
        acc[mt] = __builtin_amdgcn_wmma_f32_16x16x32_bf16(
            false, a, false, bbf, (short)0, acc[mt], false, false);
      }
    }
    __syncthreads();
  }

#pragma unroll
  for (int mt = 0; mt < 2; ++mt)
#pragma unroll
    for (int r = 0; r < 8; ++r) {
      int row = row0 + mt * 16 + r + 8 * hs;
      int col = n0 + wv * 16 + ln;
      size_t idx = (size_t)row * Dd + col;
      y[idx] = acc[mt][r] + bo[col] + x[idx];
    }
}

// ---------------------------------------------------------------------------
// Kernel 4: LayerNorm over rows of 1024.  grid = B*S, block = 256.
// ---------------------------------------------------------------------------
__global__ __launch_bounds__(256) void ln_kernel(
    const float* __restrict__ y, const float* __restrict__ w,
    const float* __restrict__ bseg, float* __restrict__ out) {
  int row = blockIdx.x;
  int c0 = threadIdx.x * 4;
  const float* yr = y + (size_t)row * Dd;
  float4 vv = *(const float4*)(yr + c0);
  float s = vv.x + vv.y + vv.z + vv.w;
  float s2 = vv.x * vv.x + vv.y * vv.y + vv.z * vv.z + vv.w * vv.w;
#pragma unroll
  for (int off = 1; off < 32; off <<= 1) {
    s  += __shfl_xor(s, off, 32);
    s2 += __shfl_xor(s2, off, 32);
  }
  __shared__ float red[2][8];
  int wvi = threadIdx.x >> 5, lane = threadIdx.x & 31;
  if (lane == 0) { red[0][wvi] = s; red[1][wvi] = s2; }
  __syncthreads();
  float ts = 0.f, ts2 = 0.f;
#pragma unroll
  for (int i = 0; i < 8; ++i) { ts += red[0][i]; ts2 += red[1][i]; }
  float mu = ts / Dd;
  float var = ts2 / Dd - mu * mu;
  float inv = rsqrtf(var + 1e-5f);
  float4 w4 = *(const float4*)(w + c0);
  float4 b4 = *(const float4*)(bseg + c0);
  float4 o4;
  o4.x = (vv.x - mu) * inv * w4.x + b4.x;
  o4.y = (vv.y - mu) * inv * w4.y + b4.y;
  o4.z = (vv.z - mu) * inv * w4.z + b4.z;
  o4.w = (vv.w - mu) * inv * w4.w + b4.w;
  *(float4*)(out + (size_t)row * Dd + c0) = o4;
}

// ---------------------------------------------------------------------------
extern "C" void kernel_launch(void* const* d_in, const int* in_sizes, int n_in,
                              void* d_out, int out_size, void* d_ws, size_t ws_size,
                              hipStream_t stream) {
  (void)in_sizes; (void)n_in; (void)out_size; (void)ws_size;
  const float* x   = (const float*)d_in[0];
  const float* Wq  = (const float*)d_in[1];
  const float* bq  = (const float*)d_in[2];
  const float* Wk  = (const float*)d_in[3];
  const float* bk  = (const float*)d_in[4];
  const float* Wv  = (const float*)d_in[5];
  const float* bv  = (const float*)d_in[6];
  const float* Wo  = (const float*)d_in[7];
  const float* bo  = (const float*)d_in[8];
  const float* lnw = (const float*)d_in[9];
  const float* lnb = (const float*)d_in[10];
  float* out = (float*)d_out;

  char* ws = (char*)d_ws;
  const size_t QB = (size_t)Bb * Hh * Ss * DH * sizeof(__bf16);   // 16 MB each
  __bf16* q   = (__bf16*)(ws);
  __bf16* k   = (__bf16*)(ws + QB);
  __bf16* v   = (__bf16*)(ws + 2 * QB);
  __bf16* cc  = (__bf16*)(ws + 3 * QB);
  float*  y   = (float*)(ws + 4 * QB);
  __bf16* woh = (__bf16*)(ws + 4 * QB + (size_t)Bb * Ss * Dd * sizeof(float));

  cvt_kernel<<<(Dd * Dd) / (256 * 8), 256, 0, stream>>>(Wo, woh, Dd * Dd);
  qkv_kernel<<<dim3(Ss / 32, Bb * Hh, 3), 256, 0, stream>>>(
      x, Wq, bq, Wk, bk, Wv, bv, q, k, v);
  attn_kernel<<<dim3(Ss / 128, Bb * Hh), 256, 0, stream>>>(q, k, v, cc);
  oproj_kernel<<<dim3((Bb * Ss) / 32, Dd / 128), 256, 0, stream>>>(cc, woh, bo, x, y);
  ln_kernel<<<Bb * Ss, 256, 0, stream>>>(y, lnw, lnb, out);
}
